// Attention_22909355556874
// MI455X (gfx1250) — matrix operand order repaired and checked by reading, not compiled, not run on hardware
//
#include <hip/hip_runtime.h>

#define DEV __device__ __forceinline__

constexpr int S_LEN  = 2048;
constexpr int DIM    = 2880;
constexpr int NH     = 64;
constexpr int NKV    = 8;
constexpr int HD     = 64;
constexpr int HDALL  = NH * HD;       // 4096
constexpr float ATT_SCALE = 0.125f;   // 1/sqrt(64)

typedef __attribute__((ext_vector_type(16))) __bf16 v16bf;
typedef __attribute__((ext_vector_type(8)))  float  v8f;

DEV __bf16 f2bf(float f) { return (__bf16)f; }   // native cvt (v_cvt_*bf16_f32)

DEV v8f wmma_bf16(v16bf a, v16bf b, v8f c) {
    return __builtin_amdgcn_wmma_f32_16x16x32_bf16(
        /*neg_a=*/false, a, /*neg_b=*/false, b,
        /*c_mod=*/(short)0, c, /*reuse_a=*/false, /*reuse_b=*/false);
}

// ---- fragment loaders (ISA 7.12.2 layouts, wave32) ----
// A: 16(M) x 32(K). Lane L: row = L&15; VGPR v holds K = 2v + (v>=4?8:0) + (L>=16?8:0)
DEV v16bf load_a_f32(const float* p, int ld, int lane) {
    int r0 = lane & 15, hi = (lane >> 4) & 1;
    const float* row = p + r0 * ld + hi * 8;
    v16bf a;
#pragma unroll
    for (int v = 0; v < 8; ++v) {
        int k = 2 * v + (v >= 4 ? 8 : 0);
        a[2 * v]     = f2bf(row[k]);
        a[2 * v + 1] = f2bf(row[k + 1]);
    }
    return a;
}
DEV v16bf load_a_bf16(const __bf16* p, int ld, int lane) {
    int r0 = lane & 15, hi = (lane >> 4) & 1;
    const __bf16* row = p + r0 * ld + hi * 8;
    v16bf a;
#pragma unroll
    for (int v = 0; v < 8; ++v) {
        int k = 2 * v + (v >= 4 ? 8 : 0);
        a[2 * v]     = row[k];
        a[2 * v + 1] = row[k + 1];
    }
    return a;
}
// B: 32(K) x 16(N), element(k,n) = p[n*ld + k]. Lane L: n = L&15; K = 2v + (v>=4?8:0) + (L>=16?8:0)
DEV v16bf load_b_f32(const float* p, int ld, int lane) {
    int r0 = lane & 15, hi = (lane >> 4) & 1;
    const float* row = p + r0 * ld + hi * 8;
    v16bf b;
#pragma unroll
    for (int v = 0; v < 8; ++v) {
        int k = 2 * v + (v >= 4 ? 8 : 0);
        b[2 * v]     = f2bf(row[k]);
        b[2 * v + 1] = f2bf(row[k + 1]);
    }
    return b;
}
DEV v16bf load_b_bf16(const __bf16* p, int ld, int lane) {
    int r0 = lane & 15, hi = (lane >> 4) & 1;
    const __bf16* row = p + r0 * ld + hi * 8;
    v16bf b;
#pragma unroll
    for (int v = 0; v < 8; ++v) {
        int k = 2 * v + (v >= 4 ? 8 : 0);
        b[2 * v]     = row[k];
        b[2 * v + 1] = row[k + 1];
    }
    return b;
}

// ================= Q/K projection + RoPE =================
// grid: (S/128, nheads), block: 128 (4 waves). Wave: 32 rows x 64 cols (one head).
// dst: bf16 [nheads][S][64]
__global__ void proj_rope_kernel(const float* __restrict__ x,
                                 const float* __restrict__ w,
                                 const float* __restrict__ bias,
                                 const float* __restrict__ rope,
                                 __bf16* __restrict__ dst) {
    int lane = threadIdx.x & 31, wid = threadIdx.x >> 5;
    int r0 = lane & 15, hi = (lane >> 4) & 1;
    int qbase = blockIdx.x * 128 + wid * 32;
    int h = blockIdx.y;
    int colbase = h * 64;

    v8f d[2][4] = {{{}, {}, {}, {}}, {{}, {}, {}, {}}};
    for (int k = 0; k < DIM; k += 32) {
        v16bf a0 = load_a_f32(x + (size_t)qbase * DIM + k, DIM, lane);
        v16bf a1 = load_a_f32(x + (size_t)(qbase + 16) * DIM + k, DIM, lane);
#pragma unroll
        for (int g = 0; g < 4; ++g) {
            v16bf b = load_b_f32(w + (size_t)(colbase + g * 16) * DIM + k, DIM, lane);
            d[0][g] = wmma_bf16(a0, b, d[0][g]);
            d[1][g] = wmma_bf16(a1, b, d[1][g]);
        }
    }
    float bb[4];
#pragma unroll
    for (int g = 0; g < 4; ++g) bb[g] = bias[colbase + g * 16 + r0];

    __bf16* dh = dst + (size_t)h * S_LEN * HD;
#pragma unroll
    for (int mi = 0; mi < 2; ++mi) {
#pragma unroll
        for (int r = 0; r < 8; ++r) {
            int s = qbase + mi * 16 + r + hi * 8;
            // RoPE pairs (j, j+32): j = g*16 + r0 for g in {0,1}; halves are frags g and g+2
            {
                int j = r0;
                float c  = rope[s * HD + j];
                float sn = rope[s * HD + 32 + j];
                float t1 = d[mi][0][r] + bb[0], t2 = d[mi][2][r] + bb[2];
                dh[(size_t)s * HD + j]      = f2bf(t1 * c - t2 * sn);
                dh[(size_t)s * HD + 32 + j] = f2bf(t2 * c + t1 * sn);
            }
            {
                int j = 16 + r0;
                float c  = rope[s * HD + j];
                float sn = rope[s * HD + 32 + j];
                float t1 = d[mi][1][r] + bb[1], t2 = d[mi][3][r] + bb[3];
                dh[(size_t)s * HD + j]      = f2bf(t1 * c - t2 * sn);
                dh[(size_t)s * HD + 32 + j] = f2bf(t2 * c + t1 * sn);
            }
        }
    }
}

// ================= V projection, transposed store =================
// dst: bf16 [NKV][64][S]  (Vt[kv][d][s]);  grid: (S/128, NKV)
__global__ void proj_v_kernel(const float* __restrict__ x,
                              const float* __restrict__ w,
                              const float* __restrict__ bias,
                              __bf16* __restrict__ dst) {
    int lane = threadIdx.x & 31, wid = threadIdx.x >> 5;
    int r0 = lane & 15, hi = (lane >> 4) & 1;
    int qbase = blockIdx.x * 128 + wid * 32;
    int h = blockIdx.y;
    int colbase = h * 64;

    v8f d[2][4] = {{{}, {}, {}, {}}, {{}, {}, {}, {}}};
    for (int k = 0; k < DIM; k += 32) {
        v16bf a0 = load_a_f32(x + (size_t)qbase * DIM + k, DIM, lane);
        v16bf a1 = load_a_f32(x + (size_t)(qbase + 16) * DIM + k, DIM, lane);
#pragma unroll
        for (int g = 0; g < 4; ++g) {
            v16bf b = load_b_f32(w + (size_t)(colbase + g * 16) * DIM + k, DIM, lane);
            d[0][g] = wmma_bf16(a0, b, d[0][g]);
            d[1][g] = wmma_bf16(a1, b, d[1][g]);
        }
    }
#pragma unroll
    for (int g = 0; g < 4; ++g) {
        float bb = bias[colbase + g * 16 + r0];
        int dd = g * 16 + r0;
#pragma unroll
        for (int mi = 0; mi < 2; ++mi)
#pragma unroll
            for (int r = 0; r < 8; ++r) {
                int s = qbase + mi * 16 + r + hi * 8;
                dst[((size_t)h * HD + dd) * S_LEN + s] = f2bf(d[mi][g][r] + bb);
            }
    }
}

// ================= Flash attention (causal, GQA, sink) =================
// grid: (S/64, NH), block 128. Wave: 16-query tile of one head.
// Computes S^T = K·Q^T so the score D-frag lane/key mapping equals the A-frag
// layout needed for P·V (pure in-lane repack, no LDS round-trip).
__global__ void flash_attn_kernel(const __bf16* __restrict__ Qr,
                                  const __bf16* __restrict__ Kr,
                                  const __bf16* __restrict__ Vt,
                                  const float* __restrict__ sinks,
                                  __bf16* __restrict__ attn) {
    int lane = threadIdx.x & 31, wid = threadIdx.x >> 5;
    int r0 = lane & 15, hi = (lane >> 4) & 1;
    int h = blockIdx.y;
    int kv = h >> 3;                        // REP = 8
    int qbase = blockIdx.x * 64 + wid * 16;

    const __bf16* Qh = Qr + (size_t)h  * S_LEN * HD;
    const __bf16* Kh = Kr + (size_t)kv * S_LEN * HD;
    const __bf16* Vh = Vt + (size_t)kv * HD * S_LEN;

    v16bf bq0 = load_b_bf16(Qh + (size_t)qbase * HD +  0, HD, lane);
    v16bf bq1 = load_b_bf16(Qh + (size_t)qbase * HD + 32, HD, lane);

    float m = -1e30f, l = 0.0f;
    v8f acc[4] = {{}, {}, {}, {}};
    int q = qbase + r0;                     // lane's query (score layout: N = query)
    int nchunk = (qbase + 16 + 31) / 32;

    for (int c = 0; c < nchunk; ++c) {
        int k0 = c * 32;
        v16bf a00 = load_a_bf16(Kh + (size_t)k0 * HD +  0, HD, lane);
        v16bf a01 = load_a_bf16(Kh + (size_t)k0 * HD + 32, HD, lane);
        v8f d0 = {};
        d0 = wmma_bf16(a00, bq0, d0);
        d0 = wmma_bf16(a01, bq1, d0);
        v16bf a10 = load_a_bf16(Kh + (size_t)(k0 + 16) * HD +  0, HD, lane);
        v16bf a11 = load_a_bf16(Kh + (size_t)(k0 + 16) * HD + 32, HD, lane);
        v8f d1 = {};
        d1 = wmma_bf16(a10, bq0, d1);
        d1 = wmma_bf16(a11, bq1, d1);

        float s0[8], s1[8];
        float cm = -1e30f;
#pragma unroll
        for (int r = 0; r < 8; ++r) {
            int key = k0 + hi * 8 + r;
            float v0 = d0[r] * ATT_SCALE; if (key > q)      v0 = -1e30f;
            float v1 = d1[r] * ATT_SCALE; if (key + 16 > q) v1 = -1e30f;
            s0[r] = v0; s1[r] = v1;
            cm = fmaxf(cm, fmaxf(v0, v1));
        }
        cm = fmaxf(cm, __shfl_xor(cm, 16));
        float mn = fmaxf(m, cm);
        float alpha = __expf(m - mn);
        float rs = 0.0f;
#pragma unroll
        for (int r = 0; r < 8; ++r) {
            s0[r] = __expf(s0[r] - mn);
            s1[r] = __expf(s1[r] - mn);
            rs += s0[r] + s1[r];
        }
        rs += __shfl_xor(rs, 16);
        l = l * alpha + rs;
        m = mn;

        // pack P into A-fragment: elems 0..7 <- s0, elems 8..15 <- s1
        v16bf ap;
#pragma unroll
        for (int r = 0; r < 8; ++r) {
            ap[r]     = f2bf(s0[r]);
            ap[8 + r] = f2bf(s1[r]);
        }
        float al[8];
#pragma unroll
        for (int r = 0; r < 8; ++r) al[r] = __shfl(alpha, r + hi * 8);
#pragma unroll
        for (int g = 0; g < 4; ++g)
#pragma unroll
            for (int r = 0; r < 8; ++r) acc[g][r] *= al[r];

#pragma unroll
        for (int g = 0; g < 4; ++g) {
            v16bf bv = load_b_bf16(Vh + (size_t)(g * 16) * S_LEN + k0, S_LEN, lane);
            acc[g] = wmma_bf16(ap, bv, acc[g]);
        }
    }

    float lse = m + __logf(l);
    float ss = 1.0f / (1.0f + __expf(-(lse - sinks[h])));
    float factor = ss / l;
    float fr[8];
#pragma unroll
    for (int r = 0; r < 8; ++r) fr[r] = __shfl(factor, r + hi * 8);

#pragma unroll
    for (int g = 0; g < 4; ++g) {
        int dd = g * 16 + r0;
#pragma unroll
        for (int r = 0; r < 8; ++r) {
            int s = qbase + r + hi * 8;
            attn[(size_t)s * HDALL + h * HD + dd] = f2bf(acc[g][r] * fr[r]);
        }
    }
}

// ================= Output projection =================
// grid: (S/128, DIM/64), block 128. Wave: 32 rows x 64 cols.
__global__ void oproj_kernel(const __bf16* __restrict__ attn,
                             const float* __restrict__ wo,
                             const float* __restrict__ bias,
                             float* __restrict__ out) {
    int lane = threadIdx.x & 31, wid = threadIdx.x >> 5;
    int r0 = lane & 15, hi = (lane >> 4) & 1;
    int qbase = blockIdx.x * 128 + wid * 32;
    int colbase = blockIdx.y * 64;

    v8f d[2][4] = {{{}, {}, {}, {}}, {{}, {}, {}, {}}};
    for (int k = 0; k < HDALL; k += 32) {
        v16bf a0 = load_a_bf16(attn + (size_t)qbase * HDALL + k, HDALL, lane);
        v16bf a1 = load_a_bf16(attn + (size_t)(qbase + 16) * HDALL + k, HDALL, lane);
#pragma unroll
        for (int g = 0; g < 4; ++g) {
            v16bf b = load_b_f32(wo + (size_t)(colbase + g * 16) * HDALL + k, HDALL, lane);
            d[0][g] = wmma_bf16(a0, b, d[0][g]);
            d[1][g] = wmma_bf16(a1, b, d[1][g]);
        }
    }
#pragma unroll
    for (int g = 0; g < 4; ++g) {
        int col = colbase + g * 16 + r0;
        float bb = bias[col];
#pragma unroll
        for (int mi = 0; mi < 2; ++mi)
#pragma unroll
            for (int r = 0; r < 8; ++r) {
                int s = qbase + mi * 16 + r + hi * 8;
                out[(size_t)s * DIM + col] = d[mi][g][r] + bb;
            }
    }
}

extern "C" void kernel_launch(void* const* d_in, const int* in_sizes, int n_in,
                              void* d_out, int out_size, void* d_ws, size_t ws_size,
                              hipStream_t stream) {
    const float* x     = (const float*)d_in[0];
    const float* rope  = (const float*)d_in[1];
    const float* wq_w  = (const float*)d_in[2];
    const float* wq_b  = (const float*)d_in[3];
    const float* wk_w  = (const float*)d_in[4];
    const float* wk_b  = (const float*)d_in[5];
    const float* wv_w  = (const float*)d_in[6];
    const float* wv_b  = (const float*)d_in[7];
    const float* wo_w  = (const float*)d_in[8];
    const float* wo_b  = (const float*)d_in[9];
    const float* sinks = (const float*)d_in[10];
    float* out = (float*)d_out;

    char* ws = (char*)d_ws;
    __bf16* Qr   = (__bf16*)(ws);                                   // 64*2048*64
    __bf16* Kr   = (__bf16*)(ws + 16777216);                        // 8*2048*64
    __bf16* Vt   = (__bf16*)(ws + 16777216 + 2097152);              // 8*64*2048
    __bf16* attn = (__bf16*)(ws + 16777216 + 2097152 + 2097152);    // 2048*4096

    dim3 blk(128);
    proj_rope_kernel<<<dim3(S_LEN / 128, NH),  blk, 0, stream>>>(x, wq_w, wq_b, rope, Qr);
    proj_rope_kernel<<<dim3(S_LEN / 128, NKV), blk, 0, stream>>>(x, wk_w, wk_b, rope, Kr);
    proj_v_kernel   <<<dim3(S_LEN / 128, NKV), blk, 0, stream>>>(x, wv_w, wv_b, Vt);
    flash_attn_kernel<<<dim3(S_LEN / 64, NH), blk, 0, stream>>>(Qr, Kr, Vt, sinks, attn);
    oproj_kernel    <<<dim3(S_LEN / 128, DIM / 64), blk, 0, stream>>>(attn, wo_w, wo_b, out);
}
